// MCQuantiles_42915313221745
// MI455X (gfx1250) — matrix operand order repaired and checked by reading, not compiled, not run on hardware
//
#include <hip/hip_runtime.h>

// ---------------------------------------------------------------------------
// MCQuantiles spiking net, CDNA5 (gfx1250) bf16-WMMA pipeline.
//
//   1. convert weights fp32 -> bf16, TRANSPOSED to [N][K] (async-copy friendly)
//   2. distAll  = te  [32768x3136] @ Wd   (WMMA, async-LDS B tiles)
//      proxAll  = se  [  512x3136] @ Wp
//      trunkAll = ee  [  512x3136] @ Wt
//   3. EMA over T for prox/trunk only (dist EMA fused into soma kernel)
//   4. soma recursion (inline d-EMA, per-element over T) -> bf16 spikes sAll
//   5. hAll  = sAll [32768x512] @ W1 + b1 (WMMA, async-LDS A+B tiles)
//   6. lif2 recursion -> bf16 spikes hsAll
//   7. outflat = hsAll [32768x512] @ W2pad[64x512]^T (WMMA)
//   8. finalize: mean over T, + b2, transpose -> out[B, LAST, S]
// ---------------------------------------------------------------------------

#define DEVINL __device__ __forceinline__

typedef __bf16 bf16_t;
typedef bf16_t v16bf __attribute__((ext_vector_type(16)));
typedef float  v8f   __attribute__((ext_vector_type(8)));
typedef unsigned int u32x8 __attribute__((ext_vector_type(8)));
typedef int v4i __attribute__((ext_vector_type(4)));

#if defined(__HIP_DEVICE_COMPILE__) && \
    __has_builtin(__builtin_amdgcn_global_load_async_to_lds_b128) && \
    __has_builtin(__builtin_amdgcn_s_wait_asynccnt)
#define HAVE_ASYNC_LDS 1
#else
#define HAVE_ASYNC_LDS 0
#endif

#if HAVE_ASYNC_LDS
typedef const __attribute__((address_space(1))) void* gas_vptr;
typedef __attribute__((address_space(3))) void*       las_vptr;
typedef __attribute__((address_space(1))) v4i*        gas_v4i;
typedef __attribute__((address_space(3))) v4i*        las_v4i;
DEVINL void async_g2l_b128(const void* g, void* l) {
    gas_vptr g1 = (gas_vptr)g;          // generic -> AS(1)
    las_vptr l3 = (las_vptr)l;          // generic -> AS(3)
    __builtin_amdgcn_global_load_async_to_lds_b128((gas_v4i)g1, (las_v4i)l3, 0, 0);
}
#endif

DEVINL unsigned short f2bf_bits(float f) {
    unsigned u = __float_as_uint(f);
    u += 0x7FFFu + ((u >> 16) & 1u);          // round-to-nearest-even
    return (unsigned short)(u >> 16);
}

// ---------------------------------------------------------------------------
// Tiled WMMA GEMM:  C[M,N] = A[M,K] * BwT[N,K]^T (+ bias[N])
//   A: fp32 (VALU-converted to bf16 while staging) or bf16 bits (ushort,
//      staged via GLOBAL_LOAD_ASYNC_TO_LDS_B128 when available)
//   BwT: bf16 bits, PRE-TRANSPOSED [N][K] row-major -> raw async tile copies.
//   M%128==0, N%64==0, K%32==0 at every call site.
//   Block: 256 threads = 8 waves; BM=128, BN=64, BK=32 (one WMMA K-step).
//   Wave (wm,wn) owns a 32x32 macro-tile = 2x2 v_wmma_f32_16x16x32_bf16.
// ---------------------------------------------------------------------------
constexpr int BM = 128, BN = 64, BK = 32;
constexpr int LPAD = 48;   // LDS row stride (halves): 96B = 16B-aligned rows,
                           // non-power-of-two dword stride vs 64 banks

template <bool A_IS_F32>
__global__ __launch_bounds__(256)
void gemm_wmma_kernel(const void* __restrict__ Aopaque,
                      const unsigned short* __restrict__ BwT,
                      float* __restrict__ C,
                      const float* __restrict__ bias,
                      int M, int N, int K)
{
    (void)M;
    __shared__ unsigned short As[BM][LPAD];
    __shared__ unsigned short Bs[BN][LPAD];

    const int tid  = threadIdx.x;
    const int wave = tid >> 5;
    const int lane = tid & 31;
    const int kh   = lane >> 4;      // which 16-lane half
    const int lr   = lane & 15;
    const int wm   = wave & 3;       // 4 row-groups of 32
    const int wn   = wave >> 2;      // 2 col-groups of 32

    const int m0 = blockIdx.y * BM;
    const int n0 = blockIdx.x * BN;

    v8f acc[2][2];
#pragma unroll
    for (int i = 0; i < 2; ++i)
#pragma unroll
        for (int j = 0; j < 2; ++j)
            acc[i][j] = v8f{0.f, 0.f, 0.f, 0.f, 0.f, 0.f, 0.f, 0.f};

    const float*          Af = (const float*)Aopaque;
    const unsigned short* Ah = (const unsigned short*)Aopaque;

    for (int k0 = 0; k0 < K; k0 += BK) {
        // prefetch next A K-tile (global_prefetch_b8)
        if (k0 + BK < K) {
            if (A_IS_F32) __builtin_prefetch(Af + (size_t)(m0 + (tid >> 3)) * K + k0 + BK, 0, 0);
            else          __builtin_prefetch(Ah + (size_t)(m0 + (tid >> 1)) * K + k0 + BK, 0, 0);
        }

        // ---- stage A tile (bf16 bits) into LDS ----
        if (A_IS_F32) {
            const int r = tid >> 3;          // 0..31
            const int c = (tid & 7) * 4;     // 0..28
#pragma unroll
            for (int p = 0; p < 4; ++p) {
                const float4 v = *(const float4*)(Af + (size_t)(m0 + r + p * 32) * K + k0 + c);
                unsigned int* d32 = (unsigned int*)&As[r + p * 32][c];
                d32[0] = (unsigned)f2bf_bits(v.x) | ((unsigned)f2bf_bits(v.y) << 16);
                d32[1] = (unsigned)f2bf_bits(v.z) | ((unsigned)f2bf_bits(v.w) << 16);
            }
        } else {
            const int r = tid >> 1;          // 0..127
            const int c = (tid & 1) * 16;    // 0 or 16
            const unsigned short* g = Ah + (size_t)(m0 + r) * K + k0 + c;
#if HAVE_ASYNC_LDS
            async_g2l_b128(g,     &As[r][c]);
            async_g2l_b128(g + 8, &As[r][c + 8]);
#else
            const uint4 v = *(const uint4*)g;
            unsigned int* d32 = (unsigned int*)&As[r][c];
            d32[0] = v.x; d32[1] = v.y; d32[2] = v.z; d32[3] = v.w;
#endif
        }

        // ---- stage B tile (already [n][k] in global) into LDS ----
        {
            const int n  = tid >> 2;         // 0..63
            const int ch = (tid & 3) * 8;    // 0,8,16,24
            const unsigned short* g = BwT + (size_t)(n0 + n) * K + k0 + ch;
#if HAVE_ASYNC_LDS
            async_g2l_b128(g, &Bs[n][ch]);
#else
            const uint4 v = *(const uint4*)g;
            unsigned int* d32 = (unsigned int*)&Bs[n][ch];
            d32[0] = v.x; d32[1] = v.y; d32[2] = v.z; d32[3] = v.w;
#endif
        }

#if HAVE_ASYNC_LDS
        __builtin_amdgcn_s_wait_asynccnt(0);
#endif
        __syncthreads();

        // ---- build fragments per CDNA5 16-bit A/B layouts ----
        v16bf afr[2], bfr[2];
        const int kb = kh * 8;   // A: lanes 0-15 -> K {0..7,16..23}; lanes 16-31 -> +8
#pragma unroll
        for (int tm = 0; tm < 2; ++tm) {
            const unsigned short* Ap = &As[wm * 32 + tm * 16 + lr][0];
            u32x8 av;
#pragma unroll
            for (int j = 0; j < 4; ++j) {
                av[j]     = *(const unsigned int*)(Ap + kb + 2 * j);
                av[j + 4] = *(const unsigned int*)(Ap + kb + 16 + 2 * j);
            }
            afr[tm] = __builtin_bit_cast(v16bf, av);
        }
#pragma unroll
        for (int tn = 0; tn < 2; ++tn) {
            // B: col = lane&15, lanes 0-15 K {0..7,16..23}; lanes 16-31 -> +8
            const unsigned short* Bp = &Bs[wn * 32 + tn * 16 + lr][kh * 8];
            u32x8 bv;
#pragma unroll
            for (int j = 0; j < 4; ++j) {
                bv[j]     = *(const unsigned int*)(Bp + 2 * j);
                bv[j + 4] = *(const unsigned int*)(Bp + 16 + 2 * j);
            }
            bfr[tn] = __builtin_bit_cast(v16bf, bv);
        }

#pragma unroll
        for (int tm = 0; tm < 2; ++tm)
#pragma unroll
            for (int tn = 0; tn < 2; ++tn)
                acc[tm][tn] = __builtin_amdgcn_wmma_f32_16x16x32_bf16(
                    /*neg_a=*/false, afr[tm], /*neg_b=*/false, bfr[tn],
                    /*c_mod=*/(short)0, acc[tm][tn],
                    /*reuse_a=*/false, /*reuse_b=*/false);

        __syncthreads();
    }

    // ---- epilogue: D layout -> row = i + 8*kh, col = lane&15 ----
#pragma unroll
    for (int tm = 0; tm < 2; ++tm) {
#pragma unroll
        for (int tn = 0; tn < 2; ++tn) {
            const int col = n0 + wn * 32 + tn * 16 + lr;
            const float bv = bias ? bias[col] : 0.f;
#pragma unroll
            for (int i = 0; i < 8; ++i) {
                const int row = m0 + wm * 32 + tm * 16 + kh * 8 + i;
                C[(size_t)row * N + col] = acc[tm][tn][i] + bv;
            }
        }
    }
}

// ---------------------------------------------------------------------------
// Elementwise kernels
// ---------------------------------------------------------------------------

// fp32 [K][Nsrc] -> bf16-bits [Ndst][K] (transposed; rows >= Nsrc zero-filled)
__global__ void f2bfT_kernel(const float* __restrict__ src,
                             unsigned short* __restrict__ dst,
                             int K, int Nsrc, int Ndst)
{
    size_t i = (size_t)blockIdx.x * blockDim.x + threadIdx.x;
    if (i >= (size_t)Ndst * K) return;
    const int n = (int)(i / K);
    const int k = (int)(i % K);
    dst[i] = (n < Nsrc) ? f2bf_bits(src[(size_t)k * Nsrc + n]) : (unsigned short)0;
}

// In-place EMA over T with 1/tau = 0.5 (linear compartments p, t)
__global__ void ema_kernel(float* __restrict__ x, size_t perT, int T)
{
    size_t i = (size_t)blockIdx.x * blockDim.x + threadIdx.x;
    if (i >= perT) return;
    float v = 0.f;
    for (int t = 0; t < T; ++t) {
        float in = x[(size_t)t * perT + i];
        v += (in - v) * 0.5f;
        x[(size_t)t * perT + i] = v;
    }
}

// soma recursion per element over all T (inline EMA of d); emits bf16 spikes
__global__ void soma_kernel(const float* __restrict__ distRaw,
                            const float* __restrict__ proxE,
                            const float* __restrict__ trunkE,
                            unsigned short* __restrict__ sAll,
                            int Bsz, int S, int F, int T)
{
    const size_t n = (size_t)Bsz * S * F;
    size_t i = (size_t)blockIdx.x * blockDim.x + threadIdx.x;
    if (i >= n) return;
    const int f = (int)(i % F);
    const size_t row = i / F;            // b*S + s
    const int b = (int)(row / S);
    const size_t RS = (size_t)Bsz * S;
    float soma = 0.f, dst8 = 0.f;
    for (int t = 0; t < T; ++t) {
        const float p   = proxE [((size_t)t * Bsz + b) * F + f];
        const float tr  = trunkE[((size_t)t * Bsz + b) * F + f];
        const float din = distRaw[((size_t)t * RS + row) * F + f];
        dst8 += (din - dst8) * 0.5f;                 // d-compartment EMA (fused)
        soma += (p * dst8 + tr - soma) * 0.5f;
        const float s = (soma >= 0.5f) ? 1.f : 0.f;
        sAll[((size_t)t * RS + row) * F + f] = (s != 0.f) ? (unsigned short)0x3F80 : (unsigned short)0;
        soma *= (1.f - s);
    }
}

// second LIF (on h = emb@W1 + b1), emits bf16 spikes
__global__ void lif2_kernel(const float* __restrict__ hAll,
                            unsigned short* __restrict__ hsAll,
                            size_t perT, int T)
{
    size_t i = (size_t)blockIdx.x * blockDim.x + threadIdx.x;
    if (i >= perT) return;
    float mem = 0.f;
    for (int t = 0; t < T; ++t) {
        const float h = hAll[(size_t)t * perT + i];
        mem += (h - mem) * 0.5f;
        const float s = (mem >= 0.5f) ? 1.f : 0.f;
        hsAll[(size_t)t * perT + i] = (s != 0.f) ? (unsigned short)0x3F80 : (unsigned short)0;
        mem *= (1.f - s);
    }
}

// mean over T, add b2, transpose -> out[B, LAST, S]
__global__ void finalize_kernel(const float* __restrict__ outflat,
                                const float* __restrict__ b2,
                                float* __restrict__ out,
                                int Bsz, int S, int LAST, int NP, int T)
{
    const size_t n = (size_t)Bsz * LAST * S;
    size_t i = (size_t)blockIdx.x * blockDim.x + threadIdx.x;
    if (i >= n) return;
    const int sI = (int)(i % S);
    const int l  = (int)((i / S) % LAST);
    const int b  = (int)(i / ((size_t)S * LAST));
    const size_t RS  = (size_t)Bsz * S;
    const size_t row = (size_t)b * S + sI;
    float acc = 0.f;
    for (int t = 0; t < T; ++t)
        acc += outflat[((size_t)t * RS + row) * NP + l];
    out[i] = b2[l] + 0.25f * acc;
}

// ---------------------------------------------------------------------------
extern "C" void kernel_launch(void* const* d_in, const int* in_sizes, int n_in,
                              void* d_out, int out_size, void* d_ws, size_t ws_size,
                              hipStream_t stream)
{
    (void)in_sizes; (void)n_in; (void)out_size; (void)ws_size;

    const float* se = (const float*)d_in[0];   // [T,B,3136]
    const float* te = (const float*)d_in[1];   // [T,B*S,3136]
    const float* ee = (const float*)d_in[2];   // [T,B,3136]
    const float* Wp = (const float*)d_in[3];   // [3136,512]
    const float* Wd = (const float*)d_in[4];
    const float* Wt = (const float*)d_in[5];
    const float* W1 = (const float*)d_in[6];   // [512,512]
    const float* b1 = (const float*)d_in[7];   // [512]
    const float* W2 = (const float*)d_in[8];   // [512,18]
    const float* b2 = (const float*)d_in[9];   // [18]
    float* out = (float*)d_out;                // [B,18,S] fp32

    constexpr int T = 4, Bsz = 128, S = 64, F = 512, KIN = 3136, HID = 512;
    constexpr int LAST = 18, NP = 64;
    constexpr int RS = Bsz * S;        // 8192
    constexpr int MT = T * RS;         // 32768

    // workspace carve-up (256B aligned); hAll aliases distAll, hsAll aliases sAll
    char* ws = (char*)d_ws;
    size_t off = 0;
    auto carve = [&](size_t bytes) -> char* {
        char* p = ws + off;
        off = (off + bytes + 255) & ~(size_t)255;
        return p;
    };
    unsigned short* WpT  = (unsigned short*)carve((size_t)F * KIN * 2);   // [F][KIN]
    unsigned short* WdT  = (unsigned short*)carve((size_t)F * KIN * 2);
    unsigned short* WtT  = (unsigned short*)carve((size_t)F * KIN * 2);
    unsigned short* W1T  = (unsigned short*)carve((size_t)HID * HID * 2); // [HID][HID]
    unsigned short* W2T  = (unsigned short*)carve((size_t)NP * HID * 2);  // [64][HID]
    float*          distAll  = (float*)carve((size_t)MT * F * 4);
    float*          proxAll  = (float*)carve((size_t)T * Bsz * F * 4);
    float*          trunkAll = (float*)carve((size_t)T * Bsz * F * 4);
    unsigned short* sAll     = (unsigned short*)carve((size_t)MT * F * 2);
    float*          outflat  = (float*)carve((size_t)MT * NP * 4);
    float*          hAll  = distAll;           // dist consumed before h is produced
    unsigned short* hsAll = sAll;              // spikes consumed before hs is produced

    const dim3 blk(256);

    // 1. weight conversion + transpose (bf16, [N][K])
    {
        const size_t nw = (size_t)F * KIN;
        f2bfT_kernel<<<(unsigned)((nw + 255) / 256), blk, 0, stream>>>(Wp, WpT, KIN, F, F);
        f2bfT_kernel<<<(unsigned)((nw + 255) / 256), blk, 0, stream>>>(Wd, WdT, KIN, F, F);
        f2bfT_kernel<<<(unsigned)((nw + 255) / 256), blk, 0, stream>>>(Wt, WtT, KIN, F, F);
        const size_t n1 = (size_t)HID * HID;
        f2bfT_kernel<<<(unsigned)((n1 + 255) / 256), blk, 0, stream>>>(W1, W1T, HID, HID, HID);
        const size_t n2 = (size_t)NP * HID;
        f2bfT_kernel<<<(unsigned)((n2 + 255) / 256), blk, 0, stream>>>(W2, W2T, HID, LAST, NP);
    }

    // 2. batched input GEMMs (all T at once)
    gemm_wmma_kernel<true><<<dim3(F / BN, MT / BM), blk, 0, stream>>>(
        te, WdT, distAll, nullptr, MT, F, KIN);
    gemm_wmma_kernel<true><<<dim3(F / BN, (T * Bsz) / BM), blk, 0, stream>>>(
        se, WpT, proxAll, nullptr, T * Bsz, F, KIN);
    gemm_wmma_kernel<true><<<dim3(F / BN, (T * Bsz) / BM), blk, 0, stream>>>(
        ee, WtT, trunkAll, nullptr, T * Bsz, F, KIN);

    // 3. EMA over T for the small linear compartments (dist EMA fused in soma)
    {
        const size_t np = (size_t)Bsz * F;
        ema_kernel<<<(unsigned)((np + 255) / 256), blk, 0, stream>>>(proxAll, np, T);
        ema_kernel<<<(unsigned)((np + 255) / 256), blk, 0, stream>>>(trunkAll, np, T);
    }

    // 4. soma recursion -> spikes for all T
    {
        const size_t n = (size_t)RS * F;
        soma_kernel<<<(unsigned)((n + 255) / 256), blk, 0, stream>>>(
            distAll, proxAll, trunkAll, sAll, Bsz, S, F, T);
    }

    // 5. h = spikes @ W1 + b1 (all T at once)
    gemm_wmma_kernel<false><<<dim3(HID / BN, MT / BM), blk, 0, stream>>>(
        sAll, W1T, hAll, b1, MT, HID, HID);

    // 6. second LIF -> hs spikes
    {
        const size_t n = (size_t)RS * HID;
        lif2_kernel<<<(unsigned)((n + 255) / 256), blk, 0, stream>>>(hAll, hsAll, n, T);
    }

    // 7. outflat = hs @ W2pad
    gemm_wmma_kernel<false><<<dim3(NP / BN, MT / BM), blk, 0, stream>>>(
        hsAll, W2T, outflat, nullptr, MT, NP, HID);

    // 8. mean over T + bias + transpose
    {
        const size_t n = (size_t)Bsz * LAST * S;
        finalize_kernel<<<(unsigned)((n + 255) / 256), blk, 0, stream>>>(
            outflat, b2, out, Bsz, S, LAST, NP, T);
    }
}
